// MoEModel_18425409700094
// MI455X (gfx1250) — matrix operand order repaired and checked by reading, not compiled, not run on hardware
//
#include <hip/hip_runtime.h>
#include <stdint.h>

#define B_TOK 16384
#define DDIM  256
#define HDIM  256
#define ODIM  256
#define NEXP  32
#define NACT  22
#define EGRP  2               // expert-split groups (gridDim.y)
#define EPERG (NEXP / EGRP)   // 16 experts per block
#define ROWH  264             // padded LDS row stride (halfs): 256 + 8 -> bank-conflict-free
#define PANEL_HALFS (32 * ROWH)
#define NBUF  4               // LDS panel buffers (prefetch distance 3)
#define NPANEL (EPERG * 16)   // 16 panels per expert (8 W1 + 8 W2)

typedef __attribute__((ext_vector_type(16))) __bf16 v16bf;
typedef __attribute__((ext_vector_type(8)))  float  v8f;
typedef __attribute__((ext_vector_type(4)))  int    v4i;

union Frag32B {
    v16bf v;
    uint4 q[2];
    unsigned u[8];
};

// ---------- CDNA5 async global->LDS copy (ASYNCcnt), with safe fallbacks ----
#if __has_builtin(__builtin_amdgcn_global_load_async_to_lds_b128)
#define HAVE_ASYNC_LDS 1
#endif

__device__ __forceinline__ void cp16_g2l(const unsigned short* g, unsigned short* l) {
#ifdef HAVE_ASYNC_LDS
    unsigned short* gnc = const_cast<unsigned short*>(g);
    __builtin_amdgcn_global_load_async_to_lds_b128(
        (__attribute__((address_space(1))) v4i*)gnc,
        (__attribute__((address_space(3))) v4i*)l, 0, 0);
#else
    *(uint4*)l = *(const uint4*)g;   // synchronous fallback (still LDS-shared)
#endif
}

#if __has_builtin(__builtin_amdgcn_s_wait_asynccnt)
#define WAIT_ASYNC(N) __builtin_amdgcn_s_wait_asynccnt(N)
#elif defined(HAVE_ASYNC_LDS)
#define WAIT_ASYNC(N) asm volatile("s_wait_asynccnt %0" ::"n"(N) : "memory")
#else
#define WAIT_ASYNC(N) ((void)0)
#endif

__device__ __forceinline__ unsigned pack_bf16_rne(float a, float b) {
    unsigned ua = __float_as_uint(a);
    unsigned ub = __float_as_uint(b);
    ua += 0x7FFFu + ((ua >> 16) & 1u);
    ub += 0x7FFFu + ((ub >> 16) & 1u);
    return (ua >> 16) | (ub & 0xFFFF0000u);
}

// ---------------- fp32 -> bf16 (RNE) conversion ----------------
__global__ void cvt_bf16_kernel(const float* __restrict__ src,
                                unsigned short* __restrict__ dst, int n) {
    int i = blockIdx.x * blockDim.x + threadIdx.x;
    int stride = gridDim.x * blockDim.x;
    for (; i < n; i += stride) {
        unsigned u = __float_as_uint(src[i]);
        u += 0x7FFFu + ((u >> 16) & 1u);
        dst[i] = (unsigned short)(u >> 16);
    }
}

// ---------------- Boltzmann gate: one wave32 per token, lane == expert -------
__global__ void gate_kernel(const float* __restrict__ x,
                            const float* __restrict__ Wg,
                            const float* __restrict__ bg,
                            float* __restrict__ gw) {
    const int wave = threadIdx.x >> 5;
    const int lane = threadIdx.x & 31;
    const int tok  = blockIdx.x * (blockDim.x >> 5) + wave;

    const float4* xv = (const float4*)(x + (size_t)tok * DDIM);
    const float4* wv = (const float4*)(Wg + (size_t)lane * DDIM);
    float acc = 0.f;
#pragma unroll 8
    for (int d = 0; d < DDIM / 4; ++d) {
        float4 a = xv[d], w = wv[d];
        acc += a.x * w.x + a.y * w.y + a.z * w.z + a.w * w.w;
    }
    float s = (acc + bg[lane]) * 0.36787944117144233f;  // score / e

    float m = s;
#pragma unroll
    for (int off = 16; off; off >>= 1) m = fmaxf(m, __shfl_xor(m, off, 32));
    float p = __expf(s - m);
    float sum = p;
#pragma unroll
    for (int off = 16; off; off >>= 1) sum += __shfl_xor(sum, off, 32);
    float prob = p / sum;

    int rank = 0;
#pragma unroll
    for (int j = 0; j < 32; ++j) {
        float sj = __shfl(s, j, 32);
        rank += (sj > s) || (sj == s && j < lane);
    }
    float w = (rank < NACT) ? prob : 0.f;
    float ksum = w;
#pragma unroll
    for (int off = 16; off; off >>= 1) ksum += __shfl_xor(ksum, off, 32);

    gw[(size_t)tok * NEXP + lane] = w / (ksum + 1e-8f);
}

// ---------------- Main MoE kernel -------------------------------------------
// Block: 256 threads = 8 waves, each wave owns 16 tokens (128 tokens/block).
// gridDim.y splits the 32 experts into 2 groups; partials combine via f32
// atomics (2 commutative addends -> deterministic).
// Weight panels (32 rows x 256 bf16) stream through a quad-buffered LDS
// pipeline (prefetch distance 3) filled by async global->LDS copies.
__global__ __launch_bounds__(256) void moe_kernel(
    const unsigned short* __restrict__ xb,   // bf16 [B, D]
    const unsigned short* __restrict__ w1b,  // bf16 [E, H, D]
    const unsigned short* __restrict__ w2b,  // bf16 [E, O, H]
    const float* __restrict__ b1,            // [E, H]
    const float* __restrict__ b2,            // [E, O]
    const float* __restrict__ gw,            // [B, E]
    float* __restrict__ out)                 // [B, O] (pre-zeroed)
{
    __shared__ __align__(16) unsigned short smem[NBUF * PANEL_HALFS];

    const int tid  = threadIdx.x;
    const int wave = tid >> 5;
    const int lane = tid & 31;
    const int lo   = lane & 15;
    const int hi   = lane >> 4;
    const int e0   = blockIdx.y * EPERG;
    const int myTok = (blockIdx.x * 8 + wave) * 16 + lo;

    const v8f vzero = {0.f, 0.f, 0.f, 0.f, 0.f, 0.f, 0.f, 0.f};

    // Issue async copy of one 16KB weight panel into LDS buffer pidx & 3.
    auto issue_panel = [&](int pidx) {
        int e = e0 + (pidx >> 4);
        int q = pidx & 15;
        const unsigned short* src = (q < 8)
            ? (w1b + ((size_t)e << 16) + (q << 13))
            : (w2b + ((size_t)e << 16) + ((q - 8) << 13));
        unsigned short* dst = smem + (pidx & (NBUF - 1)) * PANEL_HALFS;
#pragma unroll
        for (int j = 0; j < 4; ++j) {
            int c = tid + 256 * j;                         // 16B chunk id
            cp16_g2l(src + 8 * c, dst + (c >> 5) * ROWH + (c & 31) * 8);
        }
    };
    // Wait so that panel pidx is complete (allow newer panels to stay in flight).
    auto wait_panel = [&](int pidx) {
        int rem = NPANEL - 1 - pidx;   // panels issued beyond pidx
        if (rem >= 2) { WAIT_ASYNC(8); }
        else if (rem == 1) { WAIT_ASYNC(4); }
        else { WAIT_ASYNC(0); }
    };

    // Preload x as GEMM1 B-frags (columns = tokens, K over D): reused everywhere.
    Frag32B xf[8];
    {
        const uint4* xrow = (const uint4*)(xb + (size_t)myTok * DDIM + hi * 8);
#pragma unroll
        for (int k = 0; k < 8; ++k) {
            xf[k].q[0] = xrow[4 * k + 0];
            xf[k].q[1] = xrow[4 * k + 2];
        }
    }

    v8f acc[16];
#pragma unroll
    for (int t = 0; t < 16; ++t) acc[t] = vzero;

    issue_panel(0);
    issue_panel(1);
    issue_panel(2);

#pragma unroll 1
    for (int ee = 0; ee < EPERG; ++ee) {
        const int e = e0 + ee;
        const float wgt = gw[(size_t)myTok * NEXP + e];
        const float* b1e = b1 + e * HDIM;
        const float* b2e = b2 + e * ODIM;

        Frag32B hB[8];

        // ---- GEMM1 panels q=0..7: hT tile pair + bias + ReLU -> hB[q] ------
#pragma unroll
        for (int q = 0; q < 8; ++q) {
            const int pidx = ee * 16 + q;
            wait_panel(pidx);
            __syncthreads();
            if (pidx + 3 < NPANEL) issue_panel(pidx + 3);
            const unsigned short* pan = smem + (pidx & (NBUF - 1)) * PANEL_HALFS;

            float c[2][8];
#pragma unroll
            for (int sub = 0; sub < 2; ++sub) {
                // hoist bias load: hidden behind the 8 WMMAs below
                const float4* bp = (const float4*)(b1e + q * 32 + sub * 16 + hi * 8);
                float4 bv0 = bp[0], bv1 = bp[1];

                v8f cc = vzero;
                const unsigned short* arow = pan + (sub * 16 + lo) * ROWH + hi * 8;
                Frag32B af0, af1;
                af0.q[0] = *(const uint4*)(arow);
                af0.q[1] = *(const uint4*)(arow + 16);
#pragma unroll
                for (int k = 0; k < 8; ++k) {
                    Frag32B& cur = (k & 1) ? af1 : af0;
                    Frag32B& nxt = (k & 1) ? af0 : af1;
                    if (k + 1 < 8) {
                        nxt.q[0] = *(const uint4*)(arow + 32 * (k + 1));
                        nxt.q[1] = *(const uint4*)(arow + 32 * (k + 1) + 16);
                    }
                    cc = __builtin_amdgcn_wmma_f32_16x16x32_bf16(
                        false, cur.v, false, xf[k].v, (short)0, cc, false, false);
                }
                const float bb[8] = {bv0.x, bv0.y, bv0.z, bv0.w,
                                     bv1.x, bv1.y, bv1.z, bv1.w};
#pragma unroll
                for (int r = 0; r < 8; ++r)
                    c[sub][r] = fmaxf(cc[r] + bb[r], 0.f);
            }
            // pack fp32 -> bf16 pairs + half-wave exchange into GEMM2 B layout
            unsigned pk0[4], pk1[4];
#pragma unroll
            for (int j = 0; j < 4; ++j) {
                pk0[j] = pack_bf16_rne(c[0][2 * j], c[0][2 * j + 1]);
                pk1[j] = pack_bf16_rne(c[1][2 * j], c[1][2 * j + 1]);
            }
#pragma unroll
            for (int j = 0; j < 4; ++j) {
                unsigned q0 = (unsigned)__shfl_xor((int)pk0[j], 16, 32);
                unsigned q1 = (unsigned)__shfl_xor((int)pk1[j], 16, 32);
                hB[q].u[j]     = hi ? q1 : pk0[j];
                hB[q].u[4 + j] = hi ? pk1[j] : q0;
            }
        }

        // ---- GEMM2 panels q=8..15: two o-tiles per panel -------------------
#pragma unroll
        for (int q2 = 0; q2 < 8; ++q2) {
            const int pidx = ee * 16 + 8 + q2;
            wait_panel(pidx);
            __syncthreads();
            if (pidx + 3 < NPANEL) issue_panel(pidx + 3);
            const unsigned short* pan = smem + (pidx & (NBUF - 1)) * PANEL_HALFS;

#pragma unroll
            for (int sub = 0; sub < 2; ++sub) {
                const int ot = q2 * 2 + sub;
                // hoist bias load: hidden behind the 8 WMMAs below
                const float4* bp = (const float4*)(b2e + ot * 16 + hi * 8);
                float4 bv0 = bp[0], bv1 = bp[1];

                v8f d2 = vzero;
                const unsigned short* arow = pan + (sub * 16 + lo) * ROWH + hi * 16;
                Frag32B af0, af1;
                af0.q[0] = *(const uint4*)(arow);
                af0.q[1] = *(const uint4*)(arow + 8);
#pragma unroll
                for (int k = 0; k < 8; ++k) {
                    Frag32B& cur = (k & 1) ? af1 : af0;
                    Frag32B& nxt = (k & 1) ? af0 : af1;
                    if (k + 1 < 8) {
                        nxt.q[0] = *(const uint4*)(arow + 32 * (k + 1));
                        nxt.q[1] = *(const uint4*)(arow + 32 * (k + 1) + 8);
                    }
                    d2 = __builtin_amdgcn_wmma_f32_16x16x32_bf16(
                        false, cur.v, false, hB[k].v, (short)0, d2, false, false);
                }
                const float bb[8] = {bv0.x, bv0.y, bv0.z, bv0.w,
                                     bv1.x, bv1.y, bv1.z, bv1.w};
#pragma unroll
                for (int r = 0; r < 8; ++r)
                    acc[ot][r] += wgt * (d2[r] + bb[r]);
            }
        }
    }

    // Combine the two expert-group partials: deterministic (2 commutative adds).
#pragma unroll
    for (int ot = 0; ot < 16; ++ot) {
        float* orow = out + (size_t)myTok * ODIM + ot * 16 + hi * 8;
#pragma unroll
        for (int r = 0; r < 8; ++r)
            __hip_atomic_fetch_add(orow + r, acc[ot][r],
                                   __ATOMIC_RELAXED, __HIP_MEMORY_SCOPE_AGENT);
    }
}

// ---------------- launch ----------------
extern "C" void kernel_launch(void* const* d_in, const int* in_sizes, int n_in,
                              void* d_out, int out_size, void* d_ws, size_t ws_size,
                              hipStream_t stream) {
    (void)in_sizes; (void)n_in; (void)ws_size;
    const float* x  = (const float*)d_in[0];
    const float* W1 = (const float*)d_in[1];
    const float* b1 = (const float*)d_in[2];
    const float* W2 = (const float*)d_in[3];
    const float* b2 = (const float*)d_in[4];
    const float* Wg = (const float*)d_in[5];
    const float* bg = (const float*)d_in[6];
    float* out = (float*)d_out;

    char* ws = (char*)d_ws;
    const size_t XB_BYTES  = (size_t)B_TOK * DDIM * 2;        // 8 MiB
    const size_t W1B_BYTES = (size_t)NEXP * HDIM * DDIM * 2;  // 4 MiB
    const size_t W2B_BYTES = (size_t)NEXP * ODIM * HDIM * 2;  // 4 MiB
    unsigned short* xb  = (unsigned short*)(ws);
    unsigned short* w1b = (unsigned short*)(ws + XB_BYTES);
    unsigned short* w2b = (unsigned short*)(ws + XB_BYTES + W1B_BYTES);
    float*          gwp = (float*)(ws + XB_BYTES + W1B_BYTES + W2B_BYTES);

    cvt_bf16_kernel<<<2048, 256, 0, stream>>>(x,  xb,  B_TOK * DDIM);
    cvt_bf16_kernel<<<2048, 256, 0, stream>>>(W1, w1b, NEXP * HDIM * DDIM);
    cvt_bf16_kernel<<<2048, 256, 0, stream>>>(W2, w2b, NEXP * ODIM * HDIM);

    gate_kernel<<<B_TOK / 8, 256, 0, stream>>>(x, Wg, bg, gwp);

    (void)hipMemsetAsync(out, 0, (size_t)out_size * sizeof(float), stream);

    dim3 grid(B_TOK / 128, EGRP);
    moe_kernel<<<grid, 256, 0, stream>>>(xb, w1b, w2b, b1, b2, gwp, out);
}
